// RNN_35888746725575
// MI455X (gfx1250) — matrix operand order repaired and checked by reading, not compiled, not run on hardware
//
#include <hip/hip_runtime.h>

// ---------------------------------------------------------------------------
// 2-layer LSTM (B=8192,T=512,IN=25,H=5) + FC(2560->24) for gfx1250 (MI455X).
//
// Roofline: x alone is 420MB -> ~18us floor @23.3TB/s. FLOPs (8.6GF) split:
//  - input projection (4.3GF) and final FC (1.0GF): batched GEMMs -> WMMA f16
//    (v_wmma_f32_16x16x32_f16, f32 accumulate)
//  - recurrence (~2.3GF): sequential over T, parallel over B=8192 -> scalar
//    VALU, 1 thread/batch-row, weights staged in LDS.
//
// Workspace layout (assumes ws_size >= ~210MB):
//   [0,       122880)                      W_fc converted to f16
//   [131072,  131072+167772160)            xg0 = x@W_ih0^T + b  (f16, [B*T,20])
//   [131072+167772160, +41943040)          h2 flat (f16, [B, T*H]=[B,2560])
// ---------------------------------------------------------------------------

#define B_   8192
#define T_   512
#define IN_  25
#define H_   5
#define G_   20      // 4*H
#define OUT_ 24
#define K2_  2560    // H*T

typedef __attribute__((ext_vector_type(16))) _Float16 v16h;
typedef __attribute__((ext_vector_type(8)))  float    v8f;
typedef __attribute__((ext_vector_type(4)))  _Float16 h4;
typedef __attribute__((ext_vector_type(8)))  _Float16 h8;

__device__ __forceinline__ float fast_sigmoid(float x) {
    return __builtin_amdgcn_rcpf(1.0f + __expf(-x));
}
__device__ __forceinline__ float fast_tanh(float x) {
    return __builtin_amdgcn_rcpf(1.0f + __expf(-2.0f * x)) * 2.0f - 1.0f;
}

// ---------------------------------------------------------------- kernel 0 --
__global__ void cvt_wfc_f16(const float* __restrict__ wfc,
                            _Float16* __restrict__ out, int n) {
    int i = blockIdx.x * blockDim.x + threadIdx.x;
    if (i < n) out[i] = (_Float16)wfc[i];
}

// ---------------------------------------------------------------- kernel 1 --
// xg0[r, g] = sum_k x[r, k] * W_ih0[g, k] + (b_ih0[g] + b_hh0[g])
// r in [0, B*T). One wave per 16-row M-tile iteration; K=25 -> single K=32
// WMMA step (padded); N=20 -> two 16-wide N-tiles. Weight (B) fragments are
// built once per wave and reused for every M-tile.
__global__ __launch_bounds__(256)
void proj0_wmma(const float* __restrict__ x, const float* __restrict__ Wih0,
                const float* __restrict__ bih0, const float* __restrict__ bhh0,
                _Float16* __restrict__ xg) {
    const int lane = threadIdx.x & 31;
    const int wid  = blockIdx.x * (blockDim.x >> 5) + (threadIdx.x >> 5);
    const int nw   = gridDim.x * (blockDim.x >> 5);
    const int nl   = lane & 15;
    const bool hi  = lane >= 16;

    // B fragments: lane l holds column n = l&15 (+16 for tile1);
// element e holds K = (lane>=16 ? 8 : 0) + (e<8 ? e : e+8).
    const bool nv1 = (16 + nl) < G_;
    const int  n1  = nv1 ? (16 + nl) : 0;      // clamp: always in-bounds load
    v16h bf0, bf1;
#pragma unroll
    for (int e = 0; e < 16; ++e) {
        int K  = (hi ? 8 : 0) + (e < 8 ? e : e + 8);
        int Kc = (K < IN_) ? K : 0;
        float w0 = Wih0[nl * IN_ + Kc];
        float w1 = Wih0[n1 * IN_ + Kc];
        bf0[e] = (K < IN_) ? (_Float16)w0 : (_Float16)0.0f;
        bf1[e] = (K < IN_ && nv1) ? (_Float16)w1 : (_Float16)0.0f;
    }
    const float bias0 = bih0[nl] + bhh0[nl];
    const float bias1 = nv1 ? (bih0[16 + nl] + bhh0[16 + nl]) : 0.0f;

    const int khalf = hi ? 8 : 0;
    const int ntiles = (B_ * T_) / 16;
    for (int tile = wid; tile < ntiles; tile += nw) {
        const long rowbase = (long)tile * 16;
        // A fragment: lane l holds row rowbase + (l&15);
        // element e holds K = khalf + (e<8 ? e : e+8).
        const long rb = (rowbase + nl) * IN_;
        v16h a;
#pragma unroll
        for (int e = 0; e < 16; ++e) {
            int K  = khalf + (e < 8 ? e : e + 8);
            int Kc = (K < IN_) ? K : 0;
            float v = x[rb + Kc];
            a[e] = (K < IN_) ? (_Float16)v : (_Float16)0.0f;
        }
        v8f c0, c1;
#pragma unroll
        for (int r = 0; r < 8; ++r) { c0[r] = bias0; c1[r] = bias1; }
        c0 = __builtin_amdgcn_wmma_f32_16x16x32_f16(false, a, false, bf0,
                                                    (short)0, c0, false, false);
        c1 = __builtin_amdgcn_wmma_f32_16x16x32_f16(false, a, false, bf1,
                                                    (short)0, c1, false, false);
        // C element r -> (M = r + 8*hi, N = lane&15)
#pragma unroll
        for (int r = 0; r < 8; ++r) {
            long row = rowbase + r + (hi ? 8 : 0);
            xg[row * G_ + nl] = (_Float16)c0[r];
            if (nv1) xg[row * G_ + 16 + nl] = (_Float16)c1[r];
        }
    }
}

// ---------------------------------------------------------------- kernel 2 --
// Sequential recurrence, both layers fused. One thread per batch row.
// 256 blocks x 32 threads so the 256 waves spread across WGPs.
__global__ __launch_bounds__(32)
void lstm_rec(const _Float16* __restrict__ xg,
              const float* __restrict__ Whh0,
              const float* __restrict__ Wih1, const float* __restrict__ Whh1,
              const float* __restrict__ bih1, const float* __restrict__ bhh1,
              _Float16* __restrict__ h2) {
    __shared__ float sWhh0[G_ * H_];
    __shared__ float sWih1[G_ * H_];
    __shared__ float sWhh1[G_ * H_];
    __shared__ float sB1[G_];
    for (int i = threadIdx.x; i < G_ * H_; i += blockDim.x) {
        sWhh0[i] = Whh0[i];
        sWih1[i] = Wih1[i];
        sWhh1[i] = Whh1[i];
    }
    for (int i = threadIdx.x; i < G_; i += blockDim.x) sB1[i] = bih1[i] + bhh1[i];
    __syncthreads();

    const int b = blockIdx.x * blockDim.x + threadIdx.x;
    if (b >= B_) return;

    float h0[H_], c0[H_], h1[H_], c1[H_];
#pragma unroll
    for (int j = 0; j < H_; ++j) { h0[j] = 0.f; c0[j] = 0.f; h1[j] = 0.f; c1[j] = 0.f; }

    const h4* px = (const h4*)(xg + (size_t)b * T_ * G_);
    _Float16* pout = h2 + (size_t)b * K2_;

    for (int t = 0; t < T_; ++t) {
        union { h4 v[5]; _Float16 s[G_]; } u;
#pragma unroll
        for (int q = 0; q < 5; ++q) u.v[q] = px[q];
        px += 5;

        // ----- layer 0: gates = xg_t + h0 @ Whh0^T  (order: i, f, g, o)
        float gates[G_];
#pragma unroll
        for (int g = 0; g < G_; ++g) {
            float acc = (float)u.s[g];
#pragma unroll
            for (int j = 0; j < H_; ++j) acc += sWhh0[g * H_ + j] * h0[j];
            gates[g] = acc;
        }
#pragma unroll
        for (int j = 0; j < H_; ++j) {
            float ig = fast_sigmoid(gates[j]);
            float fg = fast_sigmoid(gates[H_ + j]);
            float gg = fast_tanh(gates[2 * H_ + j]);
            float og = fast_sigmoid(gates[3 * H_ + j]);
            c0[j] = fg * c0[j] + ig * gg;
            h0[j] = og * fast_tanh(c0[j]);
        }

        // ----- layer 1: gates = b1 + h0 @ Wih1^T + h1 @ Whh1^T
#pragma unroll
        for (int g = 0; g < G_; ++g) {
            float acc = sB1[g];
#pragma unroll
            for (int j = 0; j < H_; ++j)
                acc += sWih1[g * H_ + j] * h0[j] + sWhh1[g * H_ + j] * h1[j];
            gates[g] = acc;
        }
#pragma unroll
        for (int j = 0; j < H_; ++j) {
            float ig = fast_sigmoid(gates[j]);
            float fg = fast_sigmoid(gates[H_ + j]);
            float gg = fast_tanh(gates[2 * H_ + j]);
            float og = fast_sigmoid(gates[3 * H_ + j]);
            c1[j] = fg * c1[j] + ig * gg;
            h1[j] = og * fast_tanh(c1[j]);
        }
#pragma unroll
        for (int j = 0; j < H_; ++j) pout[t * H_ + j] = (_Float16)h1[j];
    }
}

// ---------------------------------------------------------------- kernel 3 --
// out[B,24] = h2[B,2560] @ Wfc^T + b_fc, WMMA f16, one wave per 16-row tile,
// 80 K-steps of 16x16x32. All fragment loads are 16B-aligned b128s.
__global__ __launch_bounds__(32)
void fc_wmma(const _Float16* __restrict__ A, const _Float16* __restrict__ Wfc16,
             const float* __restrict__ bfc, float* __restrict__ out) {
    const int lane = threadIdx.x & 31;
    const int nl   = lane & 15;
    const bool hi  = lane >= 16;
    const long rowbase = (long)blockIdx.x * 16;

    const bool nv1 = (16 + nl) < OUT_;
    const int  n1  = nv1 ? (16 + nl) : 0;

    v8f c0, c1;
    const float bias0 = bfc[nl];
    const float bias1 = nv1 ? bfc[16 + nl] : 0.0f;
#pragma unroll
    for (int r = 0; r < 8; ++r) { c0[r] = bias0; c1[r] = bias1; }

    const long arow  = (rowbase + nl) * (long)K2_;
    const int  khalf = hi ? 16 : 0;
    const int  kboff = hi ? 16 : 0;

    for (int kk = 0; kk < K2_ / 32; ++kk) {
        const int kb = kk * 32;
        // A fragment
        h8 alo = *(const h8*)(A + arow + kb + khalf);
        h8 ahi = *(const h8*)(A + arow + kb + khalf + 8);
        v16h a;
#pragma unroll
        for (int e = 0; e < 8; ++e) { a[e] = alo[e]; a[e + 8] = ahi[e]; }
        // B fragments (Wfc rows are B columns)
        const long wr0 = (long)nl * K2_ + kb + kboff;
        h8 b0lo = *(const h8*)(Wfc16 + wr0);
        h8 b0hi = *(const h8*)(Wfc16 + wr0 + 8);
        const long wr1 = (long)n1 * K2_ + kb + kboff;
        h8 b1lo = *(const h8*)(Wfc16 + wr1);
        h8 b1hi = *(const h8*)(Wfc16 + wr1 + 8);
        v16h bf0, bf1;
#pragma unroll
        for (int e = 0; e < 8; ++e) {
            bf0[e]     = b0lo[e];
            bf0[e + 8] = b0hi[e];
            bf1[e]     = nv1 ? b1lo[e] : (_Float16)0.0f;
            bf1[e + 8] = nv1 ? b1hi[e] : (_Float16)0.0f;
        }
        c0 = __builtin_amdgcn_wmma_f32_16x16x32_f16(false, a, false, bf0,
                                                    (short)0, c0, false, false);
        c1 = __builtin_amdgcn_wmma_f32_16x16x32_f16(false, a, false, bf1,
                                                    (short)0, c1, false, false);
    }
#pragma unroll
    for (int r = 0; r < 8; ++r) {
        long row = rowbase + r + (hi ? 8 : 0);
        out[row * OUT_ + nl] = c0[r];
        if (nv1) out[row * OUT_ + 16 + nl] = c1[r];
    }
}

// ----------------------------------------------------------------- launch ---
extern "C" void kernel_launch(void* const* d_in, const int* in_sizes, int n_in,
                              void* d_out, int out_size, void* d_ws, size_t ws_size,
                              hipStream_t stream) {
    const float* x    = (const float*)d_in[0];
    const float* Wih0 = (const float*)d_in[1];
    const float* Whh0 = (const float*)d_in[2];
    const float* bih0 = (const float*)d_in[3];
    const float* bhh0 = (const float*)d_in[4];
    const float* Wih1 = (const float*)d_in[5];
    const float* Whh1 = (const float*)d_in[6];
    const float* bih1 = (const float*)d_in[7];
    const float* bhh1 = (const float*)d_in[8];
    const float* Wfc  = (const float*)d_in[9];
    const float* bfc  = (const float*)d_in[10];
    float* out = (float*)d_out;

    char* ws = (char*)d_ws;
    _Float16* Wfc16 = (_Float16*)ws;                                // 122,880 B
    _Float16* xg0   = (_Float16*)(ws + (size_t)131072);             // 167,772,160 B
    _Float16* h2    = (_Float16*)(ws + (size_t)131072 +
                                  (size_t)B_ * T_ * G_ * 2);        // 41,943,040 B

    hipLaunchKernelGGL(cvt_wfc_f16, dim3((OUT_ * K2_ + 255) / 256), dim3(256),
                       0, stream, Wfc, Wfc16, OUT_ * K2_);
    hipLaunchKernelGGL(proj0_wmma, dim3(1024), dim3(256), 0, stream,
                       x, Wih0, bih0, bhh0, xg0);
    hipLaunchKernelGGL(lstm_rec, dim3(B_ / 32), dim3(32), 0, stream,
                       xg0, Whh0, Wih1, Whh1, bih1, bhh1, h2);
    hipLaunchKernelGGL(fc_wmma, dim3(B_ / 16), dim3(32), 0, stream,
                       h2, Wfc16, bfc, out);
}
